// GDP_62088047231547
// MI455X (gfx1250) — matrix operand order repaired and checked by reading, not compiled, not run on hardware
//
#include <hip/hip_runtime.h>

#define B_ 2
#define N_ 4096
#define D_ 128
#define HEADS_ 4
#define K_ 2048
#define HD_ 32
#define EPS_ 1e-5f

typedef __bf16 bf16;
typedef __attribute__((__ext_vector_type__(16))) __bf16 v16bf;
typedef __attribute__((__ext_vector_type__(8))) float v8f;

// ---------------------------------------------------------------------------
// WMMA helpers (CDNA5, wave32): D = A(16x32 bf16) * B(32x16 bf16) + C(16x16 f32)
// ---------------------------------------------------------------------------
__device__ __forceinline__ v8f wmma_bf16(v16bf a, v16bf b, v8f c) {
  // (neg_a, A, neg_b, B, c_mod, C, reuse_a, reuse_b)
  return __builtin_amdgcn_wmma_f32_16x16x32_bf16(false, a, false, b, (short)0, c,
                                                 false, false);
}

// A-matrix 16x32 bf16 fragment from row-major [lda] source.
// Contiguous-k layout used here (not the ISA table order): lanes 0-15 -> M=lane, K 0..15; lanes 16-31 -> M=lane-16,
// K 16..31. Vector elem i -> K=k0+i.
__device__ __forceinline__ v16bf load_a_frag(const bf16* A, int lda, int row_base,
                                             int k_base, int lane) {
  int m = lane & 15;
  int k0 = (lane & 16) ? 16 : 0;
  const bf16* p = A + (size_t)(row_base + m) * lda + k_base;
  v16bf a;
#pragma unroll
  for (int i = 0; i < 16; ++i) a[i] = p[k0 + i];
  return a;
}

// B-matrix 32x16 fragment where B[k][n] = src[(n_base+n)*ld + k_base + k]
// (source row-major with the WMMA-K dim contiguous; used for W^T, K^T, V^T).
// Contiguous-k layout (as load_a_frag, not the ISA table order): lanes 0-15 -> K 0..15, lanes 16-31 -> K 16..31, n = lane&15.
__device__ __forceinline__ v16bf load_b_frag_rowT(const bf16* src, int ld, int k_base,
                                                  int n_base, int lane) {
  int n = lane & 15;
  int koff = (lane & 16) ? 16 : 0;
  const bf16* p = src + (size_t)(n_base + n) * ld + k_base + koff;
  v16bf b;
#pragma unroll
  for (int i = 0; i < 16; ++i) b[i] = p[i];
  return b;
}

// ---------------------------------------------------------------------------
// Elementwise f32 -> bf16 conversion
// ---------------------------------------------------------------------------
__global__ void convert_f32_bf16(const float* __restrict__ src, bf16* __restrict__ dst,
                                 int n) {
  int i = blockIdx.x * blockDim.x + threadIdx.x;
  if (i < n) dst[i] = (bf16)src[i];
}

// ---------------------------------------------------------------------------
// Farthest point sampling: 1 block per batch, coords resident in LDS (48KB),
// per-iteration argmax via wave32 shfl-xor + cross-wave LDS reduction.
// ---------------------------------------------------------------------------
__global__ __launch_bounds__(512) void fps_kernel(const float* __restrict__ coords,
                                                  int* __restrict__ idx_out) {
  __shared__ float sx[N_];
  __shared__ float sy[N_];
  __shared__ float sz[N_];
  __shared__ float rv[16];
  __shared__ int ri[16];
  __shared__ int far_s;
  int b = blockIdx.x;
  int t = threadIdx.x;
  const float* c = coords + (size_t)b * N_ * 3;
  for (int p = t; p < N_; p += 512) {
    sx[p] = c[p * 3 + 0];
    sy[p] = c[p * 3 + 1];
    sz[p] = c[p * 3 + 2];
  }
  if (t == 0) far_s = 0;
  float dist[8];
#pragma unroll
  for (int j = 0; j < 8; ++j) dist[j] = 1e10f;
  int lane = t & 31, w = t >> 5;
  __syncthreads();
  for (int it = 0; it < K_; ++it) {
    int far = far_s;
    if (t == 0) idx_out[b * K_ + it] = far;  // record carry (pre-update), like lax.scan
    float fx = sx[far], fy = sy[far], fz = sz[far];
    float bv = -1.0f;
    int bi = 0;
#pragma unroll
    for (int j = 0; j < 8; ++j) {
      int p = t + j * 512;
      float dx = sx[p] - fx, dy = sy[p] - fy, dz = sz[p] - fz;
      float d = dx * dx + dy * dy + dz * dz;
      float nd = fminf(dist[j], d);
      dist[j] = nd;
      if (nd > bv) { bv = nd; bi = p; }
    }
#pragma unroll
    for (int m = 16; m >= 1; m >>= 1) {
      float ov = __shfl_xor(bv, m, 32);
      int oi = __shfl_xor(bi, m, 32);
      if (ov > bv || (ov == bv && oi < bi)) { bv = ov; bi = oi; }
    }
    if (lane == 0) { rv[w] = bv; ri[w] = bi; }
    __syncthreads();
    if (t == 0) {
      float bbv = rv[0];
      int bbi = ri[0];
      for (int ww = 1; ww < 16; ++ww)
        if (rv[ww] > bbv || (rv[ww] == bbv && ri[ww] < bbi)) { bbv = rv[ww]; bbi = ri[ww]; }
      far_s = bbi;
    }
    __syncthreads();
  }
}

// ---------------------------------------------------------------------------
// Gather sampled rows: key_val (f32 for residual) + bf16 copy for Q projection
// ---------------------------------------------------------------------------
__global__ __launch_bounds__(128) void gather_kernel(const float* __restrict__ feats,
                                                     const int* __restrict__ idx,
                                                     float* __restrict__ kvf,
                                                     bf16* __restrict__ kvb) {
  int row = blockIdx.x;  // 0 .. B*K-1
  int b = row / K_;
  int sel = idx[row];
  const float* src = feats + ((size_t)b * N_ + sel) * D_;
  int t = threadIdx.x;
  float v = src[t];
  kvf[(size_t)row * D_ + t] = v;
  kvb[(size_t)row * D_ + t] = (bf16)v;
}

// ---------------------------------------------------------------------------
// V^T: per batch [N,D] -> [D,N] so attention B-fragments are contiguous.
// ---------------------------------------------------------------------------
__global__ __launch_bounds__(256) void transpose_v_kernel(const bf16* __restrict__ src,
                                                          bf16* __restrict__ dst) {
  __shared__ bf16 tile[32][33];
  int bx = blockIdx.x;  // N/32 tiles
  int by = blockIdx.y;  // D/32 tiles
  int b = blockIdx.z;
  int tx = threadIdx.x & 31, ty = threadIdx.x >> 5;  // ty 0..7
  const bf16* s = src + (size_t)b * N_ * D_;
#pragma unroll
  for (int j = 0; j < 32; j += 8)
    tile[ty + j][tx] = s[(size_t)(bx * 32 + ty + j) * D_ + by * 32 + tx];
  __syncthreads();
  bf16* d = dst + (size_t)b * D_ * N_;
#pragma unroll
  for (int j = 0; j < 32; j += 8)
    d[(size_t)(by * 32 + ty + j) * N_ + bx * 32 + tx] = tile[tx][ty + j];
}

// ---------------------------------------------------------------------------
// Generic projection Y = (X @ W^T + bias) * scale [+ residual]
// 128 threads = 4 waves; each wave computes a 16x128 output stripe with
// 32 v_wmma_f32_16x16x32_bf16. W staged in LDS (32KB).
// ---------------------------------------------------------------------------
__global__ __launch_bounds__(128) void proj_kernel(
    const bf16* __restrict__ A, const bf16* __restrict__ W, const float* __restrict__ bias,
    const float* __restrict__ residual, bf16* __restrict__ out_bf,
    float* __restrict__ out_f32, int rows, float scale) {
  __shared__ bf16 Wlds[D_ * D_];
  int t = threadIdx.x;
  {  // cooperative 32KB stage, 16B per load
    const uint4* wsrc = (const uint4*)W;
    uint4* wdst = (uint4*)Wlds;
#pragma unroll
    for (int i = 0; i < 16; ++i) wdst[t + i * 128] = wsrc[t + i * 128];
  }
  __syncthreads();
  int wave = t >> 5, lane = t & 31;
  int row_base = (blockIdx.x * 4 + wave) * 16;
  if (row_base >= rows) return;
  v8f acc[8] = {};
  for (int kb = 0; kb < D_; kb += 32) {
    v16bf af = load_a_frag(A, D_, row_base, kb, lane);
#pragma unroll
    for (int c = 0; c < 8; ++c) {
      v16bf bf = load_b_frag_rowT(Wlds, D_, kb, c * 16, lane);
      acc[c] = wmma_bf16(af, bf, acc[c]);
    }
  }
  int n = lane & 15, half = lane >> 4;
#pragma unroll
  for (int c = 0; c < 8; ++c) {
    int col = c * 16 + n;
    float bc = bias[col];
#pragma unroll
    for (int r = 0; r < 8; ++r) {
      size_t row = (size_t)row_base + r + 8 * half;
      float v = (acc[c][r] + bc) * scale;
      if (residual) v += residual[row * D_ + col];
      if (out_bf) out_bf[row * D_ + col] = (bf16)v;
      if (out_f32) out_f32[row * D_ + col] = v;
    }
  }
}

// ---------------------------------------------------------------------------
// Flash attention. 1 wave per (batch, head, 16-row q-tile); the 4 waves of a
// block share (batch, head), so the 32-key K/V^T tiles are staged into LDS
// once per block with double-buffered gfx1250 async DMA
// (global_load_async_to_lds_b128 / ASYNCcnt), overlapping the next tile's
// global->LDS transfer with this tile's WMMAs + online softmax.
// ---------------------------------------------------------------------------
__global__ __launch_bounds__(128) void attn_kernel(const bf16* __restrict__ Q,
                                                   const bf16* __restrict__ Km,
                                                   const bf16* __restrict__ Vt,
                                                   bf16* __restrict__ ctx) {
  __shared__ bf16 Kt0[32 * 32];
  __shared__ bf16 Kt1[32 * 32];
  __shared__ bf16 Vt0[32 * 32];
  __shared__ bf16 Vt1[32 * 32];
  __shared__ bf16 Pbuf[4][16 * 32];
  int t = threadIdx.x;
  int wave = t >> 5, lane = t & 31;
  int gid = blockIdx.x * 4 + wave;  // 0..1023; all 4 waves share (b,h)
  int qt = gid & 127;
  int h = (gid >> 7) & (HEADS_ - 1);
  int b = gid >> 9;
  const bf16* Qp = Q + (size_t)b * K_ * D_;
  const bf16* Kp = Km + (size_t)b * N_ * D_ + h * HD_;        // [key][hd], ld = D
  const bf16* Vtp = Vt + (size_t)b * D_ * N_ + (size_t)h * HD_ * N_;  // [hd][key], ld = N
  // Per-thread async-DMA slice: 32 rows x 64B, 4 x b128 segments per row.
  int row = t >> 2, seg = t & 3;
  unsigned kb0 = (unsigned)(size_t)Kt0 + row * 64 + seg * 16;
  unsigned kb1 = (unsigned)(size_t)Kt1 + row * 64 + seg * 16;
  unsigned vb0 = (unsigned)(size_t)Vt0 + row * 64 + seg * 16;
  unsigned vb1 = (unsigned)(size_t)Vt1 + row * 64 + seg * 16;

#define STAGE(kk_, lk_, lv_)                                                           \
  {                                                                                    \
    unsigned long long gK =                                                            \
        (unsigned long long)(Kp + (size_t)((kk_) + row) * D_ + seg * 8);               \
    unsigned long long gV = (unsigned long long)(Vtp + (size_t)row * N_ + (kk_) + seg * 8); \
    asm volatile("global_load_async_to_lds_b128 %0, %1, off" ::"v"(lk_), "v"(gK)       \
                 : "memory");                                                          \
    asm volatile("global_load_async_to_lds_b128 %0, %1, off" ::"v"(lv_), "v"(gV)       \
                 : "memory");                                                          \
  }

  // Q already pre-scaled by 1/sqrt(HD) in the projection.
  v16bf qf = load_a_frag(Qp, D_, qt * 16, h * HD_, lane);
  v8f acc0 = {};
  v8f acc1 = {};
  float mrun[8], lrun[8];
#pragma unroll
  for (int r = 0; r < 8; ++r) { mrun[r] = -3.0e38f; lrun[r] = 0.0f; }
  bf16* P = &Pbuf[wave][0];
  int n = lane & 15, half = lane >> 4;

  STAGE(0, kb0, vb0);
  asm volatile("s_wait_asynccnt 0x0" ::: "memory");
  __syncthreads();

  for (int kk = 0; kk < N_; kk += 32) {
    int cur = (kk >> 5) & 1;
    const bf16* Ktile = cur ? Kt1 : Kt0;
    const bf16* Vtile = cur ? Vt1 : Vt0;
    if (kk + 32 < N_) {  // kick off DMA for the next tile into the other buffer
      if (cur) STAGE(kk + 32, kb0, vb0) else STAGE(kk + 32, kb1, vb1);
      __builtin_prefetch(Kp + (size_t)(kk + 64) * D_, 0, 1);
    }
    // scores: Ktile is [key][hd] -> B[k=d][n=key]
    v16bf kf0 = load_b_frag_rowT(Ktile, 32, 0, 0, lane);
    v16bf kf1 = load_b_frag_rowT(Ktile, 32, 0, 16, lane);
    v8f zero = {};
    v8f s0 = wmma_bf16(qf, kf0, zero);
    v8f s1 = wmma_bf16(qf, kf1, zero);
    float alpha[8];
#pragma unroll
    for (int r = 0; r < 8; ++r) {
      // C layout: this lane holds row m = r + 8*(lane>>4); 16 cols across the half.
      float mx = fmaxf(s0[r], s1[r]);
      mx = fmaxf(mx, __shfl_xor(mx, 1, 16));
      mx = fmaxf(mx, __shfl_xor(mx, 2, 16));
      mx = fmaxf(mx, __shfl_xor(mx, 4, 16));
      mx = fmaxf(mx, __shfl_xor(mx, 8, 16));
      float mnew = fmaxf(mrun[r], mx);
      float a = __expf(mrun[r] - mnew);
      float p0 = __expf(s0[r] - mnew);
      float p1 = __expf(s1[r] - mnew);
      float rs = p0 + p1;
      rs += __shfl_xor(rs, 1, 16);
      rs += __shfl_xor(rs, 2, 16);
      rs += __shfl_xor(rs, 4, 16);
      rs += __shfl_xor(rs, 8, 16);
      lrun[r] = lrun[r] * a + rs;
      mrun[r] = mnew;
      alpha[r] = a;
      int m = r + 8 * half;
      P[m * 32 + n] = (bf16)p0;
      P[m * 32 + 16 + n] = (bf16)p1;
      acc0[r] *= a;
      acc1[r] *= a;
    }
    __builtin_amdgcn_wave_barrier();  // keep DS store->load order across the re-layout
    v16bf pf = load_a_frag(P, 32, 0, 0, lane);
    __builtin_amdgcn_wave_barrier();
    // P@V: Vtile is [hd][key] -> B[k=key][n=hd], contiguous along keys
    v16bf vf0 = load_b_frag_rowT(Vtile, 32, 0, 0, lane);
    v16bf vf1 = load_b_frag_rowT(Vtile, 32, 0, 16, lane);
    acc0 = wmma_bf16(pf, vf0, acc0);
    acc1 = wmma_bf16(pf, vf1, acc1);
    asm volatile("s_wait_asynccnt 0x0" ::: "memory");
    __syncthreads();  // next tile resident + everyone done reading current tile
  }
#undef STAGE
#pragma unroll
  for (int r = 0; r < 8; ++r) {
    int m = r + 8 * half;
    size_t orow = (size_t)b * K_ + qt * 16 + m;
    float inv = 1.0f / lrun[r];
    ctx[orow * D_ + h * HD_ + n] = (bf16)(acc0[r] * inv);
    ctx[orow * D_ + h * HD_ + 16 + n] = (bf16)(acc1[r] * inv);
  }
}

// ---------------------------------------------------------------------------
// Row LayerNorm over D=128: 1 block (4 waves) per row.
// ---------------------------------------------------------------------------
__global__ __launch_bounds__(128) void ln_kernel(const float* __restrict__ y,
                                                 const float* __restrict__ g,
                                                 const float* __restrict__ be,
                                                 float* __restrict__ out) {
  __shared__ float s1[4];
  __shared__ float s2[4];
  int row = blockIdx.x, t = threadIdx.x;
  int lane = t & 31, w = t >> 5;
  float v = y[(size_t)row * D_ + t];
  float s = v;
#pragma unroll
  for (int m = 16; m >= 1; m >>= 1) s += __shfl_xor(s, m, 32);
  if (lane == 0) s1[w] = s;
  __syncthreads();
  float mu = (s1[0] + s1[1] + s1[2] + s1[3]) * (1.0f / D_);
  float d = v - mu;
  float q = d * d;
#pragma unroll
  for (int m = 16; m >= 1; m >>= 1) q += __shfl_xor(q, m, 32);
  if (lane == 0) s2[w] = q;
  __syncthreads();
  float var = (s2[0] + s2[1] + s2[2] + s2[3]) * (1.0f / D_);
  out[(size_t)row * D_ + t] = d * rsqrtf(var + EPS_) * g[t] + be[t];
}

// ---------------------------------------------------------------------------
extern "C" void kernel_launch(void* const* d_in, const int* in_sizes, int n_in,
                              void* d_out, int out_size, void* d_ws, size_t ws_size,
                              hipStream_t stream) {
  (void)in_sizes; (void)n_in; (void)out_size; (void)ws_size;
  const float* feats      = (const float*)d_in[0];
  const float* coords     = (const float*)d_in[1];
  const float* in_proj_w  = (const float*)d_in[2];
  const float* in_proj_b  = (const float*)d_in[3];
  const float* out_proj_w = (const float*)d_in[4];
  const float* out_proj_b = (const float*)d_in[5];
  const float* ln_gamma   = (const float*)d_in[6];
  const float* ln_beta    = (const float*)d_in[7];
  float* out = (float*)d_out;

  char* ws = (char*)d_ws;
  size_t off = 0;
  auto take = [&](size_t bytes) -> void* {
    void* p = (void*)(ws + off);
    off += (bytes + 255) & ~(size_t)255;
    return p;
  };
  int* idx       = (int*)take((size_t)B_ * K_ * sizeof(int));
  float* kv_f32  = (float*)take((size_t)B_ * K_ * D_ * sizeof(float));
  bf16* featsbf  = (bf16*)take((size_t)B_ * N_ * D_ * sizeof(bf16));
  bf16* wqkv     = (bf16*)take((size_t)3 * D_ * D_ * sizeof(bf16));
  bf16* wo       = (bf16*)take((size_t)D_ * D_ * sizeof(bf16));
  bf16* kvbf     = (bf16*)take((size_t)B_ * K_ * D_ * sizeof(bf16));
  bf16* qbf      = (bf16*)take((size_t)B_ * K_ * D_ * sizeof(bf16));
  bf16* kbf      = (bf16*)take((size_t)B_ * N_ * D_ * sizeof(bf16));
  bf16* vbf      = (bf16*)take((size_t)B_ * N_ * D_ * sizeof(bf16));
  bf16* vtbf     = (bf16*)take((size_t)B_ * D_ * N_ * sizeof(bf16));
  bf16* ctxbf    = (bf16*)take((size_t)B_ * K_ * D_ * sizeof(bf16));
  float* ybuf    = (float*)take((size_t)B_ * K_ * D_ * sizeof(float));

  // 1) bf16 conversions of activations / weights
  int nfe = B_ * N_ * D_;
  convert_f32_bf16<<<(nfe + 255) / 256, 256, 0, stream>>>(feats, featsbf, nfe);
  convert_f32_bf16<<<(3 * D_ * D_ + 255) / 256, 256, 0, stream>>>(in_proj_w, wqkv,
                                                                  3 * D_ * D_);
  convert_f32_bf16<<<(D_ * D_ + 255) / 256, 256, 0, stream>>>(out_proj_w, wo, D_ * D_);

  // 2) farthest point sampling + gather
  fps_kernel<<<B_, 512, 0, stream>>>(coords, idx);
  gather_kernel<<<B_ * K_, 128, 0, stream>>>(feats, idx, kv_f32, kvbf);

  // 3) K/V/Q projections (Q pre-scaled by 1/sqrt(HD)), then V^T for attention
  proj_kernel<<<(B_ * N_) / 64, 128, 0, stream>>>(featsbf, wqkv + D_ * D_, in_proj_b + D_,
                                                  nullptr, kbf, nullptr, B_ * N_, 1.0f);
  proj_kernel<<<(B_ * N_) / 64, 128, 0, stream>>>(featsbf, wqkv + 2 * D_ * D_,
                                                  in_proj_b + 2 * D_, nullptr, vbf,
                                                  nullptr, B_ * N_, 1.0f);
  proj_kernel<<<(B_ * K_) / 64, 128, 0, stream>>>(kvbf, wqkv, in_proj_b, nullptr, qbf,
                                                  nullptr, B_ * K_,
                                                  0.17677669529663687f);
  dim3 tgrid(N_ / 32, D_ / 32, B_);
  transpose_v_kernel<<<tgrid, 256, 0, stream>>>(vbf, vtbf);

  // 4) flash attention: 1024 q-tiles, 4 waves/block, async-DMA K/V staging
  attn_kernel<<<(B_ * HEADS_ * (K_ / 16)) / 4, 128, 0, stream>>>(qbf, kbf, vtbf, ctxbf);

  // 5) out projection + residual, then LayerNorm
  proj_kernel<<<(B_ * K_) / 64, 128, 0, stream>>>(ctxbf, wo, out_proj_b, kv_f32, nullptr,
                                                  ybuf, B_ * K_, 1.0f);
  ln_kernel<<<B_ * K_, 128, 0, stream>>>(ybuf, ln_gamma, ln_beta, out);
}
